// AttentionSource_42185168781993
// MI455X (gfx1250) — matrix operand order repaired and checked by reading, not compiled, not run on hardware
//
#include <hip/hip_runtime.h>
#include <hip/hip_bf16.h>
#include <math.h>

// ---------------------------------------------------------------------------
// Global attention (Luong) pipeline for MI455X (gfx1250, wave32, WMMA).
//   scores = out @ ctx^T -> softmax rows -> mix = attn @ ctx
//   result = tanh([mix, out] @ W^T + b)
// All GEMM operands pre-converted to bf16 once; hot loops move bf16 tiles
// with global_load_async_to_lds_b128 (double-buffered, ASYNCcnt-tracked)
// and compute with v_wmma_f32_16x16x32_bf16 (f32 accumulate).
// ---------------------------------------------------------------------------

typedef __attribute__((ext_vector_type(16))) __bf16 v16bf;
typedef __attribute__((ext_vector_type(8)))  __bf16 v8bf;
typedef __attribute__((ext_vector_type(8)))  float  v8f;

static constexpr int Bc = 16;     // batch
static constexpr int Tc = 2048;   // target len
static constexpr int Sc = 2048;   // source len
static constexpr int Dc = 1024;   // dim

#define BM  128
#define BN  128
#define BK  32
#define BKP 40   // padded LDS row stride (bf16) = 80B, keeps 16B alignment

// ---- async DMA: 128x32 bf16 row tile (ld elems) -> LDS [128][BKP] ----------
// 128 rows * 64B = 512 16B-chunks; 256 threads issue 2 async copies each.
__device__ __forceinline__ void async_tile_rows(const __bf16* __restrict__ src,
                                                int ld, __bf16 (*dst)[BKP],
                                                int tid) {
#pragma unroll
  for (int i = 0; i < 2; ++i) {
    int idx = tid + i * 256;
    int r   = idx >> 2;                 // 4 chunks per row
    int c   = (idx & 3) << 3;           // bf16 col (8 per chunk)
    const __bf16* g = src + (size_t)r * ld + c;
    unsigned lds = (unsigned)(size_t)&dst[r][c];   // low 32 bits = LDS offset
    asm volatile("global_load_async_to_lds_b128 %0, %1, off"
                 :: "v"(lds), "v"(g) : "memory");
  }
}

__device__ __forceinline__ void wait_async_all() {
  asm volatile("s_wait_asynccnt 0x0" ::: "memory");
}

// ---- register-staged transpose load: 32x128 bf16 -> LDS dst[n][k] ----------
__device__ __forceinline__ void gload_btrans(const __bf16* __restrict__ src,
                                             int ld, v8bf (&reg)[2], int tid) {
#pragma unroll
  for (int i = 0; i < 2; ++i) {
    int idx = tid + i * 256;            // 32 rows * 16 chunks
    int r   = idx >> 4;                 // k
    int c   = (idx & 15) << 3;          // n
    reg[i] = *reinterpret_cast<const v8bf*>(src + (size_t)r * ld + c);
  }
}
__device__ __forceinline__ void store_btrans(const v8bf (&reg)[2],
                                             __bf16 (*dst)[BKP], int tid) {
#pragma unroll
  for (int i = 0; i < 2; ++i) {
    int idx = tid + i * 256;
    int r   = idx >> 4;
    int c   = (idx & 15) << 3;
#pragma unroll
    for (int j = 0; j < 8; ++j) dst[c + j][r] = reg[i][j];
  }
}

// ---- WMMA fragment loads per CDNA5 ISA 7.12.2 16-bit layouts ---------------
// A (16x32): lane L: M=L&15, g=L>>4; elems 0..7 -> K=g*8+i, 8..15 -> K=g*8+16+i-8
__device__ __forceinline__ v16bf load_afrag(const __bf16 (*As)[BKP],
                                            int row0, int lane) {
  const int lrow = lane & 15, g = lane >> 4;
  const __bf16* p = &As[row0 + lrow][g * 8];
  v8bf lo = *reinterpret_cast<const v8bf*>(p);
  v8bf hi = *reinterpret_cast<const v8bf*>(p + 16);
  return __builtin_shufflevector(lo, hi, 0, 1, 2, 3, 4, 5, 6, 7,
                                 8, 9, 10, 11, 12, 13, 14, 15);
}
// B (32x16): lane L: N=L&15, g=L>>4; elems 0..7 -> K=g*8+i, 8..15 -> K=g*8+16+i-8
__device__ __forceinline__ v16bf load_bfrag(const __bf16 (*Bs)[BKP],
                                            int col0, int lane) {
  const int lcol = lane & 15, g = lane >> 4;
  const __bf16* p = &Bs[col0 + lcol][g * 8];
  v8bf lo = *reinterpret_cast<const v8bf*>(p);
  v8bf hi = *reinterpret_cast<const v8bf*>(p + 16);
  return __builtin_shufflevector(lo, hi, 0, 1, 2, 3, 4, 5, 6, 7,
                                 8, 9, 10, 11, 12, 13, 14, 15);
}

__device__ __forceinline__ void wmma_step(const __bf16 (*As)[BKP],
                                          const __bf16 (*Bs)[BKP],
                                          v8f (&acc)[4][2],
                                          int wm, int wn, int lane) {
  v16bf af[4];
  v16bf bf[2];
#pragma unroll
  for (int mt = 0; mt < 4; ++mt) af[mt] = load_afrag(As, wm * 64 + mt * 16, lane);
#pragma unroll
  for (int nt = 0; nt < 2; ++nt) bf[nt] = load_bfrag(Bs, wn * 32 + nt * 16, lane);
#pragma unroll
  for (int mt = 0; mt < 4; ++mt)
#pragma unroll
    for (int nt = 0; nt < 2; ++nt)
      acc[mt][nt] = __builtin_amdgcn_wmma_f32_16x16x32_bf16(
          false, af[mt], false, bf[nt], (short)0, acc[mt][nt], false, false);
}

// ---------------------------------------------------------------------------
// K0: elementwise fp32 -> bf16 (sizes are multiples of 1024)
// ---------------------------------------------------------------------------
__global__ __launch_bounds__(256) void cvt_bf16_kernel(
    const float* __restrict__ s, __bf16* __restrict__ d) {
  size_t i = ((size_t)blockIdx.x * 256 + threadIdx.x) * 4;
  const float4 v = *reinterpret_cast<const float4*>(s + i);
  d[i + 0] = (__bf16)v.x; d[i + 1] = (__bf16)v.y;
  d[i + 2] = (__bf16)v.z; d[i + 3] = (__bf16)v.w;
}

// ---------------------------------------------------------------------------
// K1: scores[b,t,s] = sum_d out[t,d]*ctx[s,d]; A/B tiles fully async, dbl-buf
// ---------------------------------------------------------------------------
__global__ __launch_bounds__(256) void scores_kernel(
    const __bf16* __restrict__ outp_bf, const __bf16* __restrict__ ctx_bf,
    float* __restrict__ attn) {
  __shared__ __bf16 As[2][BM][BKP];
  __shared__ __bf16 Bs[2][BN][BKP];
  const int tid = threadIdx.x, lane = tid & 31, wave = tid >> 5;
  const int wm = wave >> 2, wn = wave & 3;
  const int b = blockIdx.z, t0 = blockIdx.y * BM, s0 = blockIdx.x * BN;

  const __bf16* Ag = outp_bf + (size_t)b * Tc * Dc + (size_t)t0 * Dc;
  const __bf16* Bg = ctx_bf  + (size_t)b * Sc * Dc + (size_t)s0 * Dc;

  constexpr int NK = Dc / BK;
  async_tile_rows(Ag, Dc, As[0], tid);
  async_tile_rows(Bg, Dc, Bs[0], tid);
  wait_async_all();
  __syncthreads();

  v8f acc[4][2] = {};
  for (int it = 0; it < NK; ++it) {
    const int cur = it & 1;
    if (it + 1 < NK) {
      async_tile_rows(Ag + (it + 1) * BK, Dc, As[cur ^ 1], tid);
      async_tile_rows(Bg + (it + 1) * BK, Dc, Bs[cur ^ 1], tid);
    }
    wmma_step(As[cur], Bs[cur], acc, wm, wn, lane);
    if (it + 1 < NK) wait_async_all();
    __syncthreads();
  }

  float* C = attn + (size_t)b * Tc * Sc;
  const int g = lane >> 4, lrow = lane & 15;
#pragma unroll
  for (int mt = 0; mt < 4; ++mt)
#pragma unroll
    for (int nt = 0; nt < 2; ++nt)
#pragma unroll
      for (int r = 0; r < 8; ++r) {
        int row = t0 + wm * 64 + mt * 16 + g * 8 + r;
        int col = s0 + wn * 32 + nt * 16 + lrow;
        C[(size_t)row * Sc + col] = acc[mt][nt][r];
      }
}

// ---------------------------------------------------------------------------
// K2: row softmax over S (wave per row); writes normalized fp32 + bf16 copy
// ---------------------------------------------------------------------------
__global__ __launch_bounds__(256) void softmax_kernel(
    float* __restrict__ attn, __bf16* __restrict__ attn_bf) {
  const int lane = threadIdx.x & 31, wave = threadIdx.x >> 5;
  const size_t row = (size_t)blockIdx.x * 8 + wave;
  float*  p  = attn    + row * Sc;
  __bf16* pb = attn_bf + row * Sc;

  float v[64];
  float mx = -INFINITY;
#pragma unroll
  for (int i = 0; i < 64; ++i) {
    v[i] = p[lane + i * 32];
    mx = fmaxf(mx, v[i]);
  }
#pragma unroll
  for (int off = 16; off > 0; off >>= 1)
    mx = fmaxf(mx, __shfl_xor(mx, off, 32));

  float sum = 0.f;
#pragma unroll
  for (int i = 0; i < 64; ++i) {
    v[i] = __expf(v[i] - mx);
    sum += v[i];
  }
#pragma unroll
  for (int off = 16; off > 0; off >>= 1)
    sum += __shfl_xor(sum, off, 32);

  const float inv = 1.0f / sum;
#pragma unroll
  for (int i = 0; i < 64; ++i) {
    float r = v[i] * inv;
    p[lane + i * 32]  = r;
    pb[lane + i * 32] = (__bf16)r;
  }
}

// ---------------------------------------------------------------------------
// K3: mix[b,t,d] = sum_s attn[t,s]*ctx[s,d]. A async; B register-staged
// transpose (loads before WMMA, LDS stores after -> latency behind MMAs).
// Stores mix as bf16 (it only feeds the next bf16 GEMM).
// ---------------------------------------------------------------------------
__global__ __launch_bounds__(256) void mix_kernel(
    const __bf16* __restrict__ attn_bf, const __bf16* __restrict__ ctx_bf,
    __bf16* __restrict__ mix_bf) {
  __shared__ __bf16 As[2][BM][BKP];
  __shared__ __bf16 Bs[2][BN][BKP];
  const int tid = threadIdx.x, lane = tid & 31, wave = tid >> 5;
  const int wm = wave >> 2, wn = wave & 3;
  const int b = blockIdx.z, t0 = blockIdx.y * BM, d0 = blockIdx.x * BN;

  const __bf16* Ag = attn_bf + (size_t)b * Tc * Sc + (size_t)t0 * Sc;
  const __bf16* Bg = ctx_bf  + (size_t)b * Sc * Dc + d0;

  constexpr int NK = Sc / BK;
  v8bf breg[2];
  async_tile_rows(Ag, Sc, As[0], tid);
  gload_btrans(Bg, Dc, breg, tid);
  store_btrans(breg, Bs[0], tid);
  wait_async_all();
  __syncthreads();

  v8f acc[4][2] = {};
  for (int it = 0; it < NK; ++it) {
    const int cur = it & 1;
    if (it + 1 < NK) {
      async_tile_rows(Ag + (it + 1) * BK, Sc, As[cur ^ 1], tid);
      gload_btrans(Bg + (size_t)(it + 1) * BK * Dc, Dc, breg, tid);
    }
    wmma_step(As[cur], Bs[cur], acc, wm, wn, lane);
    if (it + 1 < NK) {
      store_btrans(breg, Bs[cur ^ 1], tid);
      wait_async_all();
    }
    __syncthreads();
  }

  __bf16* C = mix_bf + (size_t)b * Tc * Dc;
  const int g = lane >> 4, lrow = lane & 15;
#pragma unroll
  for (int mt = 0; mt < 4; ++mt)
#pragma unroll
    for (int nt = 0; nt < 2; ++nt)
#pragma unroll
      for (int r = 0; r < 8; ++r) {
        int row = t0 + wm * 64 + mt * 16 + g * 8 + r;
        int col = d0 + wn * 32 + nt * 16 + lrow;
        C[(size_t)row * Dc + col] = (__bf16)acc[mt][nt][r];
      }
}

// ---------------------------------------------------------------------------
// K4: out = tanh(mix@W[:, :D]^T + outp@W[:, D:]^T + bias); unified 64-iter
// pipeline across both K passes, everything async + double-buffered.
// ---------------------------------------------------------------------------
__global__ __launch_bounds__(256) void out_kernel(
    const __bf16* __restrict__ mix_bf, const __bf16* __restrict__ outp_bf,
    const __bf16* __restrict__ W_bf, const float* __restrict__ bias,
    float* __restrict__ out) {
  __shared__ __bf16 As[2][BM][BKP];
  __shared__ __bf16 Bs[2][BN][BKP];
  const int tid = threadIdx.x, lane = tid & 31, wave = tid >> 5;
  const int wm = wave >> 2, wn = wave & 3;
  const int b = blockIdx.z, t0 = blockIdx.y * BM, d0 = blockIdx.x * BN;

  const __bf16* A1 = mix_bf  + (size_t)b * Tc * Dc + (size_t)t0 * Dc;
  const __bf16* A2 = outp_bf + (size_t)b * Tc * Dc + (size_t)t0 * Dc;
  const __bf16* Wt = W_bf + (size_t)d0 * 2 * Dc;   // rows d0.., ld = 2D

  constexpr int NK = 2 * Dc / BK;  // 64: first 32 over mix, next 32 over outp
  auto a_ptr = [&](int it) {
    return (it < 32 ? A1 : A2) + (it & 31) * BK;
  };
  auto b_ptr = [&](int it) {
    return Wt + (it < 32 ? 0 : Dc) + (it & 31) * BK;
  };

  async_tile_rows(a_ptr(0), Dc, As[0], tid);
  async_tile_rows(b_ptr(0), 2 * Dc, Bs[0], tid);
  wait_async_all();
  __syncthreads();

  v8f acc[4][2] = {};
  for (int it = 0; it < NK; ++it) {
    const int cur = it & 1;
    if (it + 1 < NK) {
      async_tile_rows(a_ptr(it + 1), Dc, As[cur ^ 1], tid);
      async_tile_rows(b_ptr(it + 1), 2 * Dc, Bs[cur ^ 1], tid);
    }
    wmma_step(As[cur], Bs[cur], acc, wm, wn, lane);
    if (it + 1 < NK) wait_async_all();
    __syncthreads();
  }

  float* C = out + (size_t)b * Tc * Dc;
  const int g = lane >> 4, lrow = lane & 15;
#pragma unroll
  for (int mt = 0; mt < 4; ++mt)
#pragma unroll
    for (int nt = 0; nt < 2; ++nt) {
      int col = d0 + wn * 32 + nt * 16 + lrow;
      float bv = bias[col];
#pragma unroll
      for (int r = 0; r < 8; ++r) {
        int row = t0 + wm * 64 + mt * 16 + g * 8 + r;
        C[(size_t)row * Dc + col] = tanhf(acc[mt][nt][r] + bv);
      }
    }
}

// ---------------------------------------------------------------------------
extern "C" void kernel_launch(void* const* d_in, const int* in_sizes, int n_in,
                              void* d_out, int out_size, void* d_ws,
                              size_t ws_size, hipStream_t stream) {
  (void)in_sizes; (void)n_in; (void)out_size; (void)ws_size;
  const float* outp = (const float*)d_in[0];  // [B,T,D]
  const float* ctx  = (const float*)d_in[1];  // [B,S,D]
  const float* W    = (const float*)d_in[2];  // [D,2D]
  const float* bias = (const float*)d_in[3];  // [D]

  float* out  = (float*)d_out;                    // [B,T,D]
  float* attn = out + (size_t)Bc * Tc * Dc;       // [B,T,S]

  // bf16 workspace layout
  __bf16* outp_bf = (__bf16*)d_ws;                          // B*T*D
  __bf16* ctx_bf  = outp_bf + (size_t)Bc * Tc * Dc;         // B*S*D
  __bf16* W_bf    = ctx_bf  + (size_t)Bc * Sc * Dc;         // D*2D
  __bf16* attn_bf = W_bf    + (size_t)Dc * 2 * Dc;          // B*T*S
  __bf16* mix_bf  = attn_bf + (size_t)Bc * Tc * Sc;         // B*T*D

  const dim3 blk(256);
  const size_t nOut = (size_t)Bc * Tc * Dc;
  const size_t nCtx = (size_t)Bc * Sc * Dc;
  const size_t nW   = (size_t)Dc * 2 * Dc;
  cvt_bf16_kernel<<<dim3((unsigned)(nOut / 1024)), blk, 0, stream>>>(outp, outp_bf);
  cvt_bf16_kernel<<<dim3((unsigned)(nCtx / 1024)), blk, 0, stream>>>(ctx, ctx_bf);
  cvt_bf16_kernel<<<dim3((unsigned)(nW   / 1024)), blk, 0, stream>>>(W, W_bf);

  scores_kernel<<<dim3(Sc / BN, Tc / BM, Bc), blk, 0, stream>>>(outp_bf, ctx_bf, attn);
  softmax_kernel<<<dim3(Bc * Tc / 8), blk, 0, stream>>>(attn, attn_bf);
  mix_kernel<<<dim3(Dc / BN, Tc / BM, Bc), blk, 0, stream>>>(attn_bf, ctx_bf, mix_bf);
  out_kernel<<<dim3(Dc / BN, Tc / BM, Bc), blk, 0, stream>>>(mix_bf, outp_bf, W_bf, bias, out);
}
